// NeuromorphicConceptExplainer_78125455114741
// MI455X (gfx1250) — 16-bit and 8-bit matrix operand order repaired and checked by reading, compiled, not run on hardware
//
#include <hip/hip_runtime.h>
#include <hip/hip_bf16.h>


#define R_TOK   8192
#define DMODEL  1024
#define DFF     4096
#define SEQ     512
#define NHEAD   16
#define HDIM    64

typedef __attribute__((ext_vector_type(16))) _Float16 v16h;
typedef __attribute__((ext_vector_type(8)))  _Float16 v8h;
typedef __attribute__((ext_vector_type(8)))  float    v8f;
typedef __attribute__((ext_vector_type(8)))  int      v8i;
typedef __attribute__((ext_vector_type(4)))  int      v4i;
typedef __attribute__((ext_vector_type(2)))  int      v2i;

__device__ __forceinline__ float wave_max(float v) {
#pragma unroll
  for (int o = 16; o > 0; o >>= 1) v = fmaxf(v, __shfl_xor(v, o, 32));
  return v;
}
__device__ __forceinline__ float wave_sum(float v) {
#pragma unroll
  for (int o = 16; o > 0; o >>= 1) v += __shfl_xor(v, o, 32);
  return v;
}

__global__ void cvt_f32_f16(const float* __restrict__ in,
                            _Float16* __restrict__ out, int n) {
  for (int i = blockIdx.x * blockDim.x + threadIdx.x; i < n;
       i += gridDim.x * blockDim.x)
    out[i] = (_Float16)in[i];
}

__global__ void __launch_bounds__(256)
gemm_spike_f16(const _Float16* __restrict__ A, const _Float16* __restrict__ W,
               const float* __restrict__ bias, int R, int C, int K,
               _Float16* __restrict__ out_h, signed char* __restrict__ out_i8,
               _Float16* __restrict__ out_vT, int S) {
  const int lane = threadIdx.x & 31;
  const int wave = threadIdx.x >> 5;
  const int tiles_n = C >> 6;
  const int tile = blockIdx.x * (blockDim.x >> 5) + wave;
  if (tile >= (R >> 5) * tiles_n) return;
  const int tm = tile / tiles_n, tn = tile - tm * tiles_n;
  const int half = lane >> 4, mn = lane & 15;

  const _Float16* arow0 = A + (size_t)(tm * 32 + mn) * K;
  const _Float16* arow1 = A + (size_t)(tm * 32 + 16 + mn) * K;
  const _Float16* wr[4];
#pragma unroll
  for (int nt = 0; nt < 4; ++nt)
    wr[nt] = W + (size_t)(tn * 64 + nt * 16 + mn) * K;

  v8f acc[8] = {};
  for (int kb = 0; kb < K; kb += 32) {
    __builtin_prefetch(arow0 + kb + 128, 0, 3);
    __builtin_prefetch(arow1 + kb + 128, 0, 3);
    v16h a[2], bt[4];
    {
      v8h lo0 = *(const v8h*)(arow0 + kb + half * 8);
      v8h hi0 = *(const v8h*)(arow0 + kb + 16 + half * 8);
      v8h lo1 = *(const v8h*)(arow1 + kb + half * 8);
      v8h hi1 = *(const v8h*)(arow1 + kb + 16 + half * 8);
#pragma unroll
      for (int i = 0; i < 8; ++i) {
        a[0][i] = lo0[i]; a[0][i + 8] = hi0[i];
        a[1][i] = lo1[i]; a[1][i + 8] = hi1[i];
      }
    }
#pragma unroll
    for (int nt = 0; nt < 4; ++nt) {
      v8h lo = *(const v8h*)(wr[nt] + kb + half * 8);
      v8h hi = *(const v8h*)(wr[nt] + kb + 16 + half * 8);
#pragma unroll
      for (int i = 0; i < 8; ++i) { bt[nt][i] = lo[i]; bt[nt][i + 8] = hi[i]; }
    }
#pragma unroll
    for (int nt = 0; nt < 4; ++nt)
#pragma unroll
      for (int mt = 0; mt < 2; ++mt)
        acc[nt * 2 + mt] = __builtin_amdgcn_wmma_f32_16x16x32_f16(
            false, a[mt], false, bt[nt], (short)0, acc[nt * 2 + mt], false,
            false);
  }

#pragma unroll
  for (int nt = 0; nt < 4; ++nt) {
    const int col = tn * 64 + nt * 16 + mn;
    const float bval = bias[col];
#pragma unroll
    for (int mt = 0; mt < 2; ++mt) {
#pragma unroll
      for (int j = 0; j < 8; ++j) {
        const int m = mt * 16 + j + half * 8;
        const int row = tm * 32 + m;
        const float cur = acc[nt * 2 + mt][j] + bval;
        const float spike = (cur >= 1.0f) ? 1.0f : 0.0f;
        if (out_h)  out_h[(size_t)row * C + col] = (_Float16)spike;
        if (out_i8) out_i8[(size_t)row * C + col] = (signed char)spike;
        if (out_vT) {
          const int h = col >> 6, d = col & 63;
          const int b = row / S, s = row - b * S;
          out_vT[(((size_t)(b * (C >> 6) + h)) * HDIM + d) * S + s] =
              (_Float16)spike;
        }
      }
    }
  }
}

__global__ void __launch_bounds__(128)
attn_fused(const signed char* __restrict__ q8, const signed char* __restrict__ k8,
           const _Float16* __restrict__ vT, _Float16* __restrict__ attn_out) {
  __shared__ _Float16 smem[4 * 16 * SEQ];
  const int lane = threadIdx.x & 31, wave = threadIdx.x >> 5;
  const int gid = blockIdx.x * 4 + wave;
  const int qb = gid & 31;
  const int h  = (gid >> 5) & (NHEAD - 1);
  const int b  = gid >> 9;
  const int half = lane >> 4, mn = lane & 15;
  _Float16* srow = smem + wave * 16 * SEQ;

  const signed char* qrow =
      q8 + ((size_t)(b * SEQ + qb * 16 + mn)) * DMODEL + h * HDIM;
  v8i qa;
#pragma unroll
  for (int p = 0; p < 4; ++p) {
    v2i d = *(const v2i*)(qrow + (p >> 1) * 32 + half * 16 + (p & 1) * 8);
    qa[2 * p] = d.x; qa[2 * p + 1] = d.y;
  }

  for (int ct = 0; ct < 32; ++ct) {
    const signed char* krow =
        k8 + ((size_t)(b * SEQ + ct * 16 + mn)) * DMODEL + h * HDIM;
    v4i klo = *(const v4i*)(krow + half * 16);
    v4i khi = *(const v4i*)(krow + 32 + half * 16);
    v8i kv;
#pragma unroll
    for (int i = 0; i < 4; ++i) { kv[i] = klo[i]; kv[i + 4] = khi[i]; }
    v8i ci = {};
    ci = __builtin_amdgcn_wmma_i32_16x16x64_iu8(false, qa, false, kv, ci,
                                                false, false);
#pragma unroll
    for (int j = 0; j < 8; ++j) {
      const int m = j + half * 8;
      srow[m * SEQ + ct * 16 + mn] = (_Float16)((float)ci[j] * 0.125f);
    }
  }
  __syncthreads();

  for (int r = 0; r < 16; ++r) {
    _Float16* row = srow + r * SEQ;
    float mx = -1e30f;
    for (int c = lane; c < SEQ; c += 32) mx = fmaxf(mx, (float)row[c]);
    mx = wave_max(mx);
    float sum = 0.f;
    for (int c = lane; c < SEQ; c += 32) sum += __expf((float)row[c] - mx);
    sum = wave_sum(sum);
    const float inv = 1.0f / sum;
    for (int c = lane; c < SEQ; c += 32)
      row[c] = (_Float16)(__expf((float)row[c] - mx) * inv);
  }
  __syncthreads();

#pragma unroll
  for (int nt = 0; nt < 4; ++nt) {
    const _Float16* vrow =
        vT + (((size_t)(b * NHEAD + h)) * HDIM + nt * 16 + mn) * SEQ;
    v8f acc = {};
    for (int kc = 0; kc < 16; ++kc) {
      const _Float16* ar = srow + mn * SEQ + kc * 32 + half * 8;
      v8h a_lo = *(const v8h*)ar;
      v8h a_hi = *(const v8h*)(ar + 16);
      v8h b_lo = *(const v8h*)(vrow + kc * 32 + half * 8);
      v8h b_hi = *(const v8h*)(vrow + kc * 32 + 16 + half * 8);
      v16h av, bv;
#pragma unroll
      for (int i = 0; i < 8; ++i) {
        av[i] = a_lo[i]; av[i + 8] = a_hi[i];
        bv[i] = b_lo[i]; bv[i + 8] = b_hi[i];
      }
      acc = __builtin_amdgcn_wmma_f32_16x16x32_f16(false, av, false, bv,
                                                   (short)0, acc, false, false);
    }
#pragma unroll
    for (int j = 0; j < 8; ++j) {
      const int m = j + half * 8;
      const int s = qb * 16 + m;
      attn_out[((size_t)(b * SEQ + s)) * DMODEL + h * HDIM + nt * 16 + mn] =
          (_Float16)acc[j];
    }
  }
}

__global__ void __launch_bounds__(256)
residual_ln(const float* __restrict__ xin, const _Float16* __restrict__ spk,
            const float* __restrict__ gamma, const float* __restrict__ beta,
            float* __restrict__ y, _Float16* __restrict__ yh) {
  __shared__ float red[16];
  const int row = blockIdx.x, tid = threadIdx.x;
  const float* xr = xin + (size_t)row * DMODEL;
  const _Float16* sr = spk + (size_t)row * DMODEL;
  float v[4], s = 0.f, s2 = 0.f;
#pragma unroll
  for (int i = 0; i < 4; ++i) {
    const int c = tid + 256 * i;
    const float t = xr[c] + (float)sr[c];
    v[i] = t; s += t; s2 += t * t;
  }
#pragma unroll
  for (int o = 16; o > 0; o >>= 1) {
    s += __shfl_xor(s, o, 32); s2 += __shfl_xor(s2, o, 32);
  }
  const int w = tid >> 5, lane = tid & 31;
  if (lane == 0) { red[w] = s; red[8 + w] = s2; }
  __syncthreads();
  if (tid == 0) {
    float a = 0.f, bsq = 0.f;
#pragma unroll
    for (int i = 0; i < 8; ++i) { a += red[i]; bsq += red[8 + i]; }
    red[0] = a; red[8] = bsq;
  }
  __syncthreads();
  const float mean = red[0] * (1.0f / DMODEL);
  const float var  = red[8] * (1.0f / DMODEL) - mean * mean;
  const float inv  = rsqrtf(var + 1e-5f);
#pragma unroll
  for (int i = 0; i < 4; ++i) {
    const int c = tid + 256 * i;
    const float o = (v[i] - mean) * inv * gamma[c] + beta[c];
    y[(size_t)row * DMODEL + c] = o;
    if (yh) yh[(size_t)row * DMODEL + c] = (_Float16)o;
  }
}

extern "C" void kernel_launch(void* const* d_in, const int* in_sizes, int n_in,
                              void* d_out, int out_size, void* d_ws,
                              size_t ws_size, hipStream_t stream) {
  const float* x   = (const float*)d_in[0];
  const float* Wq  = (const float*)d_in[1];  const float* bq  = (const float*)d_in[2];
  const float* Wk  = (const float*)d_in[3];  const float* bk  = (const float*)d_in[4];
  const float* Wv  = (const float*)d_in[5];  const float* bv  = (const float*)d_in[6];
  const float* Wo  = (const float*)d_in[7];  const float* bo  = (const float*)d_in[8];
  const float* Wf1 = (const float*)d_in[9];  const float* bf1 = (const float*)d_in[10];
  const float* Wf2 = (const float*)d_in[11]; const float* bf2 = (const float*)d_in[12];
  const float* g1  = (const float*)d_in[13]; const float* be1 = (const float*)d_in[14];
  const float* g2  = (const float*)d_in[15]; const float* be2 = (const float*)d_in[16];

  char* p = (char*)d_ws;
  auto take = [&](size_t bytes) -> void* {
    void* r = (void*)p; p += (bytes + 255) & ~(size_t)255; return r;
  };
  _Float16* Xh   = (_Float16*)take((size_t)R_TOK * DMODEL * 2);
  _Float16* Wqh  = (_Float16*)take((size_t)DMODEL * DMODEL * 2);
  _Float16* Wkh  = (_Float16*)take((size_t)DMODEL * DMODEL * 2);
  _Float16* Wvh  = (_Float16*)take((size_t)DMODEL * DMODEL * 2);
  _Float16* Woh  = (_Float16*)take((size_t)DMODEL * DMODEL * 2);
  _Float16* Wf1h = (_Float16*)take((size_t)DFF * DMODEL * 2);
  _Float16* Wf2h = (_Float16*)take((size_t)DMODEL * DFF * 2);
  signed char* q8 = (signed char*)take((size_t)R_TOK * DMODEL);
  signed char* k8 = (signed char*)take((size_t)R_TOK * DMODEL);
  _Float16* vTh   = (_Float16*)take((size_t)R_TOK * DMODEL * 2);
  _Float16* atth  = (_Float16*)take((size_t)R_TOK * DMODEL * 2);
  _Float16* ospk  = (_Float16*)take((size_t)R_TOK * DMODEL * 2);
  float*    x1f   = (float*)   take((size_t)R_TOK * DMODEL * 4);
  _Float16* x1h   = (_Float16*)take((size_t)R_TOK * DMODEL * 2);
  _Float16* ff1h  = (_Float16*)take((size_t)R_TOK * DFF * 2);
  _Float16* ff2h  = (_Float16*)take((size_t)R_TOK * DMODEL * 2);

  cvt_f32_f16<<<1024, 256, 0, stream>>>(x,   Xh,   R_TOK * DMODEL);
  cvt_f32_f16<<<512,  256, 0, stream>>>(Wq,  Wqh,  DMODEL * DMODEL);
  cvt_f32_f16<<<512,  256, 0, stream>>>(Wk,  Wkh,  DMODEL * DMODEL);
  cvt_f32_f16<<<512,  256, 0, stream>>>(Wv,  Wvh,  DMODEL * DMODEL);
  cvt_f32_f16<<<512,  256, 0, stream>>>(Wo,  Woh,  DMODEL * DMODEL);
  cvt_f32_f16<<<1024, 256, 0, stream>>>(Wf1, Wf1h, DFF * DMODEL);
  cvt_f32_f16<<<1024, 256, 0, stream>>>(Wf2, Wf2h, DMODEL * DFF);

  const int blkD = (R_TOK / 32) * (DMODEL / 64) / 8;
  gemm_spike_f16<<<blkD, 256, 0, stream>>>(Xh, Wqh, bq, R_TOK, DMODEL, DMODEL,
                                           nullptr, q8, nullptr, SEQ);
  gemm_spike_f16<<<blkD, 256, 0, stream>>>(Xh, Wkh, bk, R_TOK, DMODEL, DMODEL,
                                           nullptr, k8, nullptr, SEQ);
  gemm_spike_f16<<<blkD, 256, 0, stream>>>(Xh, Wvh, bv, R_TOK, DMODEL, DMODEL,
                                           nullptr, nullptr, vTh, SEQ);

  attn_fused<<<2048, 128, 0, stream>>>(q8, k8, vTh, atth);

  gemm_spike_f16<<<blkD, 256, 0, stream>>>(atth, Woh, bo, R_TOK, DMODEL, DMODEL,
                                           ospk, nullptr, nullptr, SEQ);
  residual_ln<<<R_TOK, 256, 0, stream>>>(x, ospk, g1, be1, x1f, x1h);

  const int blkF1 = (R_TOK / 32) * (DFF / 64) / 8;
  gemm_spike_f16<<<blkF1, 256, 0, stream>>>(x1h, Wf1h, bf1, R_TOK, DFF, DMODEL,
                                            ff1h, nullptr, nullptr, SEQ);
  gemm_spike_f16<<<blkD, 256, 0, stream>>>(ff1h, Wf2h, bf2, R_TOK, DMODEL, DFF,
                                           ff2h, nullptr, nullptr, SEQ);
  residual_ln<<<R_TOK, 256, 0, stream>>>(x1f, ff2h, g2, be2, (float*)d_out,
                                         nullptr);
}
